// MFAM_8890582303041
// MI455X (gfx1250) — compile-verified, matrix operand order checked by reading, not run on hardware
//
#include <hip/hip_runtime.h>
#include <hip/hip_bf16.h>

typedef __attribute__((ext_vector_type(16))) __bf16 v16bf;
typedef __attribute__((ext_vector_type(8)))  float  v8f;

#define B_     16
#define C_     64
#define N_     25600      // 160*160
#define HID_   256
#define DEPTH_ 2
#define KSEL_  20480      // ceil(0.8 * 25600)

// Monotone unsigned key: larger float -> larger key
__device__ __forceinline__ unsigned mono_key(float f) {
    unsigned u = __float_as_uint(f);
    return (u & 0x80000000u) ? ~u : (u | 0x80000000u);
}

// ---------------------------------------------------------------------------
// Kernel 1: per-batch 4-round radix select of the K-th largest key.
// One block per batch. Writes threshold key T[b], #equal-keys-to-take R[b],
// and zeroes the tie counter.
// ---------------------------------------------------------------------------
__global__ void select_kernel(const float* __restrict__ prop,
                              unsigned* __restrict__ thrT,
                              int* __restrict__ thrR,
                              int* __restrict__ eqcnt) {
    __shared__ unsigned hist[256];
    __shared__ unsigned s_prefix;
    __shared__ int s_want;
    const int b = blockIdx.x;
    const float* p = prop + (size_t)b * N_;
    if (threadIdx.x == 0) { s_prefix = 0u; s_want = KSEL_; }
    __syncthreads();
    for (int round = 0; round < 4; ++round) {
        for (int i = threadIdx.x; i < 256; i += blockDim.x) hist[i] = 0u;
        __syncthreads();
        const unsigned prefix = s_prefix;
        const int shLow = 8 * (3 - round);
        for (int i = threadIdx.x; i < N_; i += blockDim.x) {
            unsigned k = mono_key(p[i]);
            bool match = (round == 0) || ((k >> (shLow + 8)) == prefix);
            if (match) atomicAdd(&hist[(k >> shLow) & 0xFFu], 1u);
        }
        __syncthreads();
        if (threadIdx.x == 0) {
            int want = s_want, cum = 0, chosen = 0;
            for (int bkt = 255; bkt >= 0; --bkt) {
                int c = (int)hist[bkt];
                if (cum + c >= want) { chosen = bkt; want -= cum; break; }
                cum += c;
            }
            s_want = want;
            s_prefix = (prefix << 8) | (unsigned)chosen;
        }
        __syncthreads();
    }
    if (threadIdx.x == 0) {
        thrT[b] = s_prefix;   // exact key of the K-th largest element
        thrR[b] = s_want;     // how many keys == T to include
        eqcnt[b] = 0;
    }
}

// ---------------------------------------------------------------------------
// Kernel 2: per-token membership mask (1.0 = in top-K).
// ---------------------------------------------------------------------------
__global__ void mask_kernel(const float* __restrict__ prop,
                            const unsigned* __restrict__ thrT,
                            const int* __restrict__ thrR,
                            int* __restrict__ eqcnt,
                            float* __restrict__ mask) {
    int idx = blockIdx.x * blockDim.x + threadIdx.x;
    if (idx >= B_ * N_) return;
    int b = idx / N_;
    unsigned k = mono_key(prop[idx]);
    unsigned T = thrT[b];
    float m = 0.f;
    if (k > T) m = 1.f;
    else if (k == T) {
        int old = atomicAdd(&eqcnt[b], 1);
        if (old < thrR[b]) m = 1.f;
    }
    mask[idx] = m;
}

// ---------------------------------------------------------------------------
// Kernel 3: pack transposed weights into per-lane WMMA A-fragment order (bf16).
//   wp1: [d][mt(16)][kc(2)][lane(32)][i(16)]   A = w1^T tiles (hidden x chan)
//   wp2: [d][mt(4)][kc(8)][lane(32)][i(16)]    A = w2^T tiles (chan x hidden)
// A-fragment element (lane,i) maps to row m = mt*16 + (lane&15),
// col K = kc*32 + ((i&8)?16:0) + 8*(lane>>4) + ((i>>1)&3)*2 + (i&1).
// ---------------------------------------------------------------------------
__global__ void pack_kernel(const float* __restrict__ w1,
                            const float* __restrict__ w2,
                            __bf16* __restrict__ wp1,
                            __bf16* __restrict__ wp2) {
    int tid = blockIdx.x * blockDim.x + threadIdx.x;
    if (tid < 32768) {
        int e = tid;
        int i = e & 15, lane = (e >> 4) & 31, kc = (e >> 9) & 1;
        int mt = (e >> 10) & 15, d = e >> 14;
        int hh = lane >> 4;
        int K = ((i & 8) ? 16 : 0) + 8 * hh + ((i >> 1) & 3) * 2 + (i & 1);
        int hid = mt * 16 + (lane & 15);
        int c = kc * 32 + K;
        wp1[e] = (__bf16)w1[(size_t)(d * C_ + c) * HID_ + hid];
    } else if (tid < 65536) {
        int e = tid - 32768;
        int i = e & 15, lane = (e >> 4) & 31, kc = (e >> 9) & 7;
        int mt = (e >> 12) & 3, d = e >> 14;
        int hh = lane >> 4;
        int K = ((i & 8) ? 16 : 0) + 8 * hh + ((i >> 1) & 3) * 2 + (i & 1);
        int cOut = mt * 16 + (lane & 15);
        int hid = kc * 32 + K;
        wp2[e] = (__bf16)w2[(size_t)(d * HID_ + hid) * C_ + cOut];
    }
}

// ---------------------------------------------------------------------------
// Kernel 4: fused masked Former stack. One wave = 16 tokens.
// D = A(W^T) x B(tokens): token index == lane&15 in both B and C/D layouts,
// and for 16-bit fragments each lane owns channel set {c : bit3(c)==lane>=16},
// so LN -> GEMM1 -> GELU -> GEMM2 stays entirely in registers.
// ---------------------------------------------------------------------------
__global__ void __launch_bounds__(256)
former_kernel(const float* __restrict__ x,
              const float* __restrict__ ln_scale,
              const float* __restrict__ ln_bias,
              const float* __restrict__ b1,
              const float* __restrict__ b2,
              const __bf16* __restrict__ wp1,
              const __bf16* __restrict__ wp2,
              const float* __restrict__ mask,
              float* __restrict__ out) {
    const int lane = threadIdx.x & 31;
    const int wv = threadIdx.x >> 5;
    const int g = blockIdx.x * 8 + wv;          // 25600 token-groups total
    const int b = g / (N_ / 16);
    const int n0 = (g % (N_ / 16)) * 16;
    const int h = lane >> 4;                    // lane half
    const int lm = lane & 15;                   // token within group

    const float* xb = x + (size_t)b * C_ * N_ + n0 + lm;
    float* ob = out + (size_t)b * C_ * N_ + n0 + lm;

    // residual t in C/D layout: t[ct][r] holds (channel = ct*16 + r + 8h, token n0+lm)
    float t[4][8];
    #pragma unroll
    for (int ct = 0; ct < 4; ++ct)
        #pragma unroll
        for (int r = 0; r < 8; ++r)
            t[ct][r] = xb[(size_t)(ct * 16 + r + 8 * h) * N_];

    const float mk = mask[(size_t)b * N_ + n0 + lm];

    #pragma unroll
    for (int d = 0; d < DEPTH_; ++d) {
        // ---- LayerNorm over 64 channels (32 local + partner half via xor-16) ----
        float s = 0.f;
        #pragma unroll
        for (int ct = 0; ct < 4; ++ct)
            #pragma unroll
            for (int r = 0; r < 8; ++r) s += t[ct][r];
        s += __shfl_xor(s, 16, 32);
        float mu = s * (1.0f / 64.0f);
        float v = 0.f;
        #pragma unroll
        for (int ct = 0; ct < 4; ++ct)
            #pragma unroll
            for (int r = 0; r < 8; ++r) { float dd = t[ct][r] - mu; v += dd * dd; }
        v += __shfl_xor(v, 16, 32);
        // single v_rsq_f32 (hardware transcendental)
        float rs = __builtin_amdgcn_rsqf(v * (1.0f / 64.0f) + 1e-5f);

        float hv[4][8];
        #pragma unroll
        for (int ct = 0; ct < 4; ++ct)
            #pragma unroll
            for (int r = 0; r < 8; ++r) {
                int c = ct * 16 + r + 8 * h;
                hv[ct][r] = (t[ct][r] - mu) * rs * ln_scale[d * C_ + c] + ln_bias[d * C_ + c];
            }

        // ---- re-fragment hv (C/D layout) into B fragments for GEMM1 ----
        // channel c = kc*32 | (i&8)<<1 | h<<3 | (i&7)  ->  ct=(kc<<1)|(i>>3), r=i&7
        v16bf bIn[2];
        #pragma unroll
        for (int kc = 0; kc < 2; ++kc)
            #pragma unroll
            for (int i = 0; i < 16; ++i)
                bIn[kc][i] = (__bf16)hv[(kc << 1) | ((i >> 3) & 1)][i & 7];

        // ---- GEMM2 accumulators init with b2 ----
        v8f accOut[4];
        #pragma unroll
        for (int mt2 = 0; mt2 < 4; ++mt2)
            #pragma unroll
            for (int r = 0; r < 8; ++r)
                accOut[mt2][r] = b2[d * C_ + mt2 * 16 + r + 8 * h];

        const __bf16* w1d = wp1 + d * 16384 + lane * 16;
        const __bf16* w2d = wp2 + d * 16384 + lane * 16;

        // ---- fused GEMM1 -> GELU -> GEMM2, streaming hidden in chunks of 32 ----
        #pragma unroll
        for (int kc2 = 0; kc2 < 8; ++kc2) {
            v8f accP, accQ;                     // hidden m-tiles 2*kc2, 2*kc2+1
            #pragma unroll
            for (int r = 0; r < 8; ++r) {
                accP[r] = b1[d * HID_ + (2 * kc2) * 16 + r + 8 * h];
                accQ[r] = b1[d * HID_ + (2 * kc2 + 1) * 16 + r + 8 * h];
            }
            #pragma unroll
            for (int kc = 0; kc < 2; ++kc) {
                v16bf aP = *(const v16bf*)(w1d + ((2 * kc2) * 2 + kc) * 512);
                v16bf aQ = *(const v16bf*)(w1d + ((2 * kc2 + 1) * 2 + kc) * 512);
                accP = __builtin_amdgcn_wmma_f32_16x16x32_bf16(
                    false, aP, false, bIn[kc], (short)0, accP, false, false);
                accQ = __builtin_amdgcn_wmma_f32_16x16x32_bf16(
                    false, aQ, false, bIn[kc], (short)0, accQ, false, false);
            }
            // GELU (tanh approx: 0.5x(1+tanh(u)) == x*sigmoid(2u)).
            // sigmoid via v_exp_f32 + single v_rcp_f32 (no IEEE div expansion).
            v16bf bAct;
            #pragma unroll
            for (int i = 0; i < 16; ++i) {
                float xv = (i & 8) ? accQ[i & 7] : accP[i & 7];
                float u = 1.5957691216f * (xv + 0.044715f * xv * xv * xv);
                float sg = __builtin_amdgcn_rcpf(1.0f + __expf(-u));
                bAct[i] = (__bf16)(xv * sg);
            }
            #pragma unroll
            for (int mt2 = 0; mt2 < 4; ++mt2) {
                v16bf a2 = *(const v16bf*)(w2d + (mt2 * 8 + kc2) * 512);
                accOut[mt2] = __builtin_amdgcn_wmma_f32_16x16x32_bf16(
                    false, a2, false, bAct, (short)0, accOut[mt2], false, false);
            }
        }

        // ---- masked residual add (mask is per-token == per-lane-column) ----
        #pragma unroll
        for (int ct = 0; ct < 4; ++ct)
            #pragma unroll
            for (int r = 0; r < 8; ++r)
                t[ct][r] += mk * accOut[ct][r];
    }

    #pragma unroll
    for (int ct = 0; ct < 4; ++ct)
        #pragma unroll
        for (int r = 0; r < 8; ++r)
            ob[(size_t)(ct * 16 + r + 8 * h) * N_] = t[ct][r];
}

// ---------------------------------------------------------------------------
extern "C" void kernel_launch(void* const* d_in, const int* in_sizes, int n_in,
                              void* d_out, int out_size, void* d_ws, size_t ws_size,
                              hipStream_t stream) {
    const float* x        = (const float*)d_in[0];
    const float* proposal = (const float*)d_in[1];
    const float* ln_scale = (const float*)d_in[2];
    const float* ln_bias  = (const float*)d_in[3];
    const float* w1       = (const float*)d_in[4];
    const float* b1       = (const float*)d_in[5];
    const float* w2       = (const float*)d_in[6];
    const float* b2       = (const float*)d_in[7];
    float* out = (float*)d_out;

    char* ws = (char*)d_ws;
    __bf16* wp1 = (__bf16*)ws;                                // 64 KB
    __bf16* wp2 = (__bf16*)(ws + 65536);                      // 64 KB
    float*  mask = (float*)(ws + 131072);                     // 1.6384 MB
    size_t off = 131072 + (size_t)B_ * N_ * sizeof(float);
    unsigned* thrT = (unsigned*)(ws + off);
    int* thrR  = (int*)(ws + off + 256);
    int* eqcnt = (int*)(ws + off + 512);

    select_kernel<<<B_, 256, 0, stream>>>(proposal, thrT, thrR, eqcnt);
    mask_kernel<<<(B_ * N_ + 255) / 256, 256, 0, stream>>>(proposal, thrT, thrR, eqcnt, mask);
    pack_kernel<<<65536 / 256, 256, 0, stream>>>(w1, w2, wp1, wp2);
    former_kernel<<<(B_ * (N_ / 16)) / 8, 256, 0, stream>>>(
        x, ln_scale, ln_bias, b1, b2, wp1, wp2, mask, out);
}
